// CategoricalModel_27728308863164
// MI455X (gfx1250) — compile-verified, matrix operand order checked by reading, not run on hardware
//
#include <hip/hip_runtime.h>

typedef __attribute__((ext_vector_type(16))) _Float16 v16h;
typedef __attribute__((ext_vector_type(8)))  float    v8f;

constexpr int kT = 128, kB = 64, kP = 2;
constexpr int kDG = 256, kDP = 256, kDE = 256, kNC = 7;
constexpr int kDMC = 1024;
constexpr int k3H = 768;          // 3 * 256 (GRU gate width)
constexpr int RNW = 16;           // waves per recurrent block (512 threads)

// ---------------------------------------------------------------------------
// WMMA fragment layout as used by this code (wave32; contiguous k order, same for A and B):
//  A 16x32 f16 frag: lane l -> row = l%16, K base = (l/16)*16, 16 contiguous halfs
//  C/D 16x16 f32 frag: VGPR r -> M = r + 8*(l/16), N = l%16
// All GEMMs are A(M,K) * W(N,K)^T with both operands row-major, so every
// fragment is a single contiguous 32-byte v16h load.
// ---------------------------------------------------------------------------
__device__ __forceinline__ float wredSum(float v) {
#pragma unroll
  for (int o = 16; o > 0; o >>= 1) v += __shfl_xor(v, o, 32);
  return v;
}
__device__ __forceinline__ float wredMax(float v) {
#pragma unroll
  for (int o = 16; o > 0; o >>= 1) v = fmaxf(v, __shfl_xor(v, o, 32));
  return v;
}
__device__ __forceinline__ float sigmoidf_(float x) { return 1.f / (1.f + __expf(-x)); }

// ---------------------------------------------------------------------------
// Generic batched WMMA GEMM: C[M,N](+bias,relu) = A[M,K] * B[N,K]^T
// Each wave owns a 16x32 output tile: one A fragment feeds two WMMAs.
// Optional f32 and/or f16 outputs.
// ---------------------------------------------------------------------------
struct GemmArgs {
  const _Float16* A; long long lda, Abatch;
  const _Float16* B; long long ldb, Bbatch;
  float* C;          long long ldc, Cbatch;
  _Float16* C16;     long long ldc16, C16batch;
  const float* bias;
  int M, N, K, relu;
};

__global__ void k_gemm(GemmArgs g) {
  const int lane = threadIdx.x & 31;
  const int nt2 = g.N >> 5;                        // N in 32-wide pairs
  const long long tiles = (long long)(g.M >> 4) * nt2;
  const long long wid = (long long)blockIdx.x * (blockDim.x >> 5) + (threadIdx.x >> 5);
  const long long nw  = (long long)gridDim.x * (blockDim.x >> 5);
  const long long bb  = blockIdx.y;
  const _Float16* A  = g.A + bb * g.Abatch;
  const _Float16* Bm = g.B + bb * g.Bbatch;
  float* C      = g.C   ? g.C   + bb * g.Cbatch   : nullptr;
  _Float16* C16 = g.C16 ? g.C16 + bb * g.C16batch : nullptr;
  const int lhalf = (lane >> 4) << 4;              // 0 or 16 (K sub-block)
  for (long long tix = wid; tix < tiles; tix += nw) {
    const int m0 = (int)(tix / nt2) << 4;
    const int n0 = (int)(tix % nt2) << 5;
    const _Float16* pa  = A  + (long long)(m0 + (lane & 15)) * g.lda + lhalf;
    const _Float16* pb0 = Bm + (long long)(n0 + (lane & 15)) * g.ldb + lhalf;
    const _Float16* pb1 = pb0 + 16 * g.ldb;
    v8f acc0 = {}, acc1 = {};
    for (int k0 = 0; k0 < g.K; k0 += 32) {
      v16h a  = *(const v16h*)(pa + k0);
      v16h b0 = *(const v16h*)(pb0 + k0);
      v16h b1 = *(const v16h*)(pb1 + k0);
      acc0 = __builtin_amdgcn_wmma_f32_16x16x32_f16(false, a, false, b0,
                                                    (short)0, acc0, false, false);
      acc1 = __builtin_amdgcn_wmma_f32_16x16x32_f16(false, a, false, b1,
                                                    (short)0, acc1, false, false);
    }
    const int col = n0 + (lane & 15);
    const int rbase = m0 + ((lane >> 4) << 3);
    const float bs0 = g.bias ? g.bias[col] : 0.f;
    const float bs1 = g.bias ? g.bias[col + 16] : 0.f;
#pragma unroll
    for (int r = 0; r < 8; ++r) {
      float v0 = acc0[r] + bs0;
      float v1 = acc1[r] + bs1;
      if (g.relu) { v0 = v0 > 0.f ? v0 : 0.f; v1 = v1 > 0.f ? v1 : 0.f; }
      const long long ro = (long long)(rbase + r);
      if (C)   { C[ro * g.ldc + col] = v0;   C[ro * g.ldc + col + 16] = v1; }
      if (C16) { C16[ro * g.ldc16 + col] = (_Float16)v0;
                 C16[ro * g.ldc16 + col + 16] = (_Float16)v1; }
    }
  }
}

// ---------------------------------------------------------------------------
// Small utility kernels
// ---------------------------------------------------------------------------
__global__ void k_counts(const float* umask, int* cb) {
  int b = threadIdx.x;
  if (b >= kB) return;
  float s = 0.f;
  for (int t = 0; t < kT; ++t) s += umask[b * kT + t];
  int c = (int)(s + 0.5f);
  if (c > kT) c = kT;
  if (c < 0) c = 0;
  cb[b] = c;
}

__global__ void k_qi(const float* qmask, const int* cb, int* qif, int* qir) {
  int i = blockIdx.x * blockDim.x + threadIdx.x;
  if (i >= kT * kB) return;
  int t = i / kB, b = i % kB;
  qif[i] = (qmask[(long long)i * kP + 1] > qmask[(long long)i * kP]) ? 1 : 0;
  int c = cb[b];
  if (t < c) {
    long long j = ((long long)(c - 1 - t) * kB + b) * kP;
    qir[i] = (qmask[j + 1] > qmask[j]) ? 1 : 0;
  } else {
    qir[i] = 0;
  }
}

__global__ void k_cvt(const float* a, _Float16* o, long long n) {
  for (long long i = (long long)blockIdx.x * blockDim.x + threadIdx.x; i < n;
       i += (long long)gridDim.x * blockDim.x)
    o[i] = (_Float16)a[i];
}

__global__ void k_reverse_f16(const float* src, _Float16* dst, const int* cb, int D) {
  long long n = (long long)kT * kB * D;
  for (long long i = (long long)blockIdx.x * blockDim.x + threadIdx.x; i < n;
       i += (long long)gridDim.x * blockDim.x) {
    int d = (int)(i % D);
    long long tb = i / D;
    int b = (int)(tb % kB), t = (int)(tb / kB);
    int c = cb[b];
    dst[i] = (t < c) ? (_Float16)src[((long long)(c - 1 - t) * kB + b) * D + d]
                     : (_Float16)0.f;
  }
}

__global__ void k_build_em(const float* ef, const float* esr, const int* cb,
                           _Float16* em) {
  long long n = (long long)kT * kB * 512;
  for (long long i = (long long)blockIdx.x * blockDim.x + threadIdx.x; i < n;
       i += (long long)gridDim.x * blockDim.x) {
    int d = (int)(i % 512);
    long long tb = i / 512;
    int b = (int)(tb % kB), t = (int)(tb / kB);
    float v;
    if (d < 256) {
      v = ef[tb * 256 + d];
    } else {
      int c = cb[b];
      v = (t < c) ? esr[((long long)(c - 1 - t) * kB + b) * 256 + (d - 256)] : 0.f;
    }
    em[i] = (_Float16)v;
  }
}

// hT[b][d][t] = h[(t*kB+b)*512 + d]   (per-batch transpose for the NT GEMMs)
__global__ void k_transpose_h(const float* h, _Float16* hT) {
  long long n = (long long)kB * 512 * kT;
  for (long long i = (long long)blockIdx.x * blockDim.x + threadIdx.x; i < n;
       i += (long long)gridDim.x * blockDim.x) {
    int t = (int)(i % kT);
    long long bd = i / kT;
    int d = (int)(bd % 512), b = (int)(bd / 512);
    hT[i] = (_Float16)h[((long long)t * kB + b) * 512 + d];
  }
}

__global__ void k_softmax_f16(const float* in, _Float16* out, int rows, int L) {
  int w = blockIdx.x * (blockDim.x >> 5) + (threadIdx.x >> 5);
  int lane = threadIdx.x & 31;
  if (w >= rows) return;
  const float* r = in + (long long)w * L;
  float mx = -1e30f;
  for (int j = lane; j < L; j += 32) mx = fmaxf(mx, r[j]);
  mx = wredMax(mx);
  float s = 0.f;
  for (int j = lane; j < L; j += 32) s += __expf(r[j] - mx);
  s = wredSum(s);
  float inv = 1.f / s;
  for (int j = lane; j < L; j += 32)
    out[(long long)w * L + j] = (_Float16)(__expf(r[j] - mx) * inv);
}

// out(T,B,1024): first half = a1 * x1, second half = a2 * x2   (a* are (B,T,512))
__global__ void k_bicombine(const float* a1f, const float* a2f,
                            const float* h1, const float* h2, float* out) {
  long long n = (long long)kT * kB * 512;
  for (long long i = (long long)blockIdx.x * blockDim.x + threadIdx.x; i < n;
       i += (long long)gridDim.x * blockDim.x) {
    int d = (int)(i % 512);
    long long tb = i / 512;
    int b = (int)(tb % kB), t = (int)(tb / kB);
    long long src = ((long long)b * kT + t) * 512 + d;
    long long o = tb * 1024;
    out[o + d]       = a1f[src] * h1[i];
    out[o + 512 + d] = a2f[src] * h2[i];
  }
}

__global__ void k_fc(const float* h_at, const float* h_va, const float* h_tv,
                     const float* ht, const float* ha, const float* hv,
                     const float* fcW, const float* fcb, float* out) {
  int w = blockIdx.x * (blockDim.x >> 5) + (threadIdx.x >> 5);
  int lane = threadIdx.x & 31;
  if (w >= kT * kB) return;
  const long long tb = w;
  float logits[kNC];
#pragma unroll 1
  for (int c = 0; c < kNC; ++c) {
    const float* wr = fcW + (long long)c * 4608;
    float acc = 0.f;
    for (int j = lane; j < 4608; j += 32) {
      float f;
      if (j < 1024)      f = h_at[tb * 1024 + j];
      else if (j < 2048) f = h_va[tb * 1024 + (j - 1024)];
      else if (j < 3072) f = h_tv[tb * 1024 + (j - 2048)];
      else if (j < 3584) f = ht[tb * 512 + (j - 3072)];
      else if (j < 4096) f = ha[tb * 512 + (j - 3584)];
      else               f = hv[tb * 512 + (j - 4096)];
      acc += wr[j] * f;
    }
    acc = wredSum(acc);
    logits[c] = acc + fcb[c];
  }
  float m = logits[0];
#pragma unroll
  for (int c = 1; c < kNC; ++c) m = fmaxf(m, logits[c]);
  float s = 0.f;
#pragma unroll
  for (int c = 0; c < kNC; ++c) s += __expf(logits[c] - m);
  float lse = m + __logf(s);
#pragma unroll
  for (int c = 0; c < kNC; ++c)
    if (lane == c) out[tb * kNC + c] = logits[c] - lse;
}

// ---------------------------------------------------------------------------
// Persistent recurrent kernel: one block (16 waves) per net, full T-step scan.
// States kept in f32 with f16 shadow copies so every GEMM operand is a
// contiguous v16h fragment load (no per-element cvt in the hot loop).
// ---------------------------------------------------------------------------
struct NetCtx {
  const _Float16 *WihG, *WhhG, *WihP, *WhhP, *WihE, *WhhE;
  const float *bihG, *bhhG, *bihP, *bhhP, *bihE, *bhhE, *attnW;
  const _Float16 *giU, *piU;   // precomputed input projections (T*B x 768) f16
  const int *qi;               // T*B speaker index
  float *g_prev, *q, *e_prev;  // f32 states: 64x256, 128x256, 64x256
  _Float16 *g_prevH, *qH, *e_prevH;          // f16 shadows of the states
  _Float16 *qselH, *qselnewH, *csumH;        // f16 GEMM inputs
  float *g_buf, *es;           // T*B*256
  float *ghG, *giQ, *ghP, *giC, *giE, *ghE;  // per-step GEMM outputs
  int dmP;                     // columns of WihP before the context part
};
struct AllCtx { NetCtx net[6]; };

// OUT[M x 768] = A(f16, M x 256) * W(f16, 768 x ldw)^T, 16x32 tile per wave
__device__ __forceinline__ void block_gemm_rc(const _Float16* A, int lda, int M,
                                              const _Float16* W, int ldw,
                                              float* C, int wid, int lane) {
  const int nt2 = k3H >> 5;                 // 24 pairs
  const int tiles = (M >> 4) * nt2;
  const int lhalf = (lane >> 4) << 4;
  for (int tix = wid; tix < tiles; tix += RNW) {
    const int m0 = (tix / nt2) << 4, n0 = (tix % nt2) << 5;
    const _Float16* pa  = A + (long long)(m0 + (lane & 15)) * lda + lhalf;
    const _Float16* pb0 = W + (long long)(n0 + (lane & 15)) * ldw + lhalf;
    const _Float16* pb1 = pb0 + 16 * (long long)ldw;
    v8f acc0 = {}, acc1 = {};
#pragma unroll
    for (int k0 = 0; k0 < 256; k0 += 32) {
      v16h a  = *(const v16h*)(pa + k0);
      v16h b0 = *(const v16h*)(pb0 + k0);
      v16h b1 = *(const v16h*)(pb1 + k0);
      acc0 = __builtin_amdgcn_wmma_f32_16x16x32_f16(false, a, false, b0,
                                                    (short)0, acc0, false, false);
      acc1 = __builtin_amdgcn_wmma_f32_16x16x32_f16(false, a, false, b1,
                                                    (short)0, acc1, false, false);
    }
    const int col = n0 + (lane & 15);
    const int rbase = m0 + ((lane >> 4) << 3);
#pragma unroll
    for (int r = 0; r < 8; ++r) {
      C[(long long)(rbase + r) * k3H + col] = acc0[r];
      C[(long long)(rbase + r) * k3H + col + 16] = acc1[r];
    }
  }
}

__global__ __launch_bounds__(512) void k_recurrent(AllCtx all) {
  NetCtx n = all.net[blockIdx.x];
  const int tid = threadIdx.x, wid = tid >> 5, lane = tid & 31;
  __shared__ float s_alpha[RNW][kT];

  for (int i = tid; i < kB * kDG; i += blockDim.x) {
    n.g_prev[i] = 0.f; n.g_prevH[i] = (_Float16)0.f;
    n.e_prev[i] = 0.f; n.e_prevH[i] = (_Float16)0.f;
  }
  for (int i = tid; i < kB * kP * kDP; i += blockDim.x) {
    n.q[i] = 0.f; n.qH[i] = (_Float16)0.f;
  }
  __syncthreads();

  for (int t = 0; t < kT; ++t) {
    const int* qit = n.qi + t * kB;

    // ---- q_sel = q[b, qi] (f16 GEMM operand) ----
    for (int i = tid; i < kB * kDP; i += blockDim.x) {
      int b = i >> 8, k = i & 255;
      n.qselH[i] = (_Float16)n.q[((b << 1) + qit[b]) * kDP + k];
    }

    // ---- attention over g_buf[0..t) -> csum (f16) ----
    for (int bi = 0; bi < kB / RNW; ++bi) {
      int b = bi * RNW + wid;
      float vals[4]; float mx = -1e30f; int c2 = 0;
      for (int s = lane; s < t; s += 32) {
        const float* gr = n.g_buf + ((long long)s * kB + b) * kDG;
        float acc = 0.f;
        for (int k = 0; k < kDG; ++k) acc += gr[k] * n.attnW[k];
        vals[c2++] = acc; mx = fmaxf(mx, acc);
      }
      mx = wredMax(mx);
      float ex[4]; float sum = 0.f; c2 = 0;
      for (int s = lane; s < t; s += 32) { float e = __expf(vals[c2] - mx); ex[c2] = e; sum += e; ++c2; }
      sum = wredSum(sum);
      float inv = (t > 0) ? 1.f / sum : 0.f;
      c2 = 0;
      for (int s = lane; s < t; s += 32) s_alpha[wid][s] = ex[c2++] * inv;
      __syncthreads();
      for (int k = lane; k < kDG; k += 32) {
        float acc = 0.f;
        for (int s = 0; s < t; ++s)
          acc += s_alpha[wid][s] * n.g_buf[((long long)s * kB + b) * kDG + k];
        n.csumH[b * kDG + k] = (_Float16)acc;
      }
      __syncthreads();
    }

    // ---- recurrent GEMMs (WMMA) ----
    block_gemm_rc(n.g_prevH, kDG, kB,      n.WhhG,         kDG,         n.ghG, wid, lane);
    block_gemm_rc(n.qselH,   kDP, kB,      n.WihG + kDMC,  kDMC + kDP,  n.giQ, wid, lane);
    block_gemm_rc(n.qH,      kDP, kB * kP, n.WhhP,         kDP,         n.ghP, wid, lane);
    block_gemm_rc(n.csumH,   kDG, kB,      n.WihP + n.dmP, n.dmP + kDG, n.giC, wid, lane);
    __syncthreads();

    // ---- g GRU combine ----
    for (int i = tid; i < kB * kDG; i += blockDim.x) {
      int b = i >> 8, k = i & 255;
      long long gio = ((long long)t * kB + b) * k3H;
      long long o = (long long)b * k3H + k;
      float ir  = (float)n.giU[gio + k]       + n.giQ[o]       + n.bihG[k];
      float iz  = (float)n.giU[gio + 256 + k] + n.giQ[o + 256] + n.bihG[k + 256];
      float in_ = (float)n.giU[gio + 512 + k] + n.giQ[o + 512] + n.bihG[k + 512];
      float hr = n.ghG[o]       + n.bhhG[k];
      float hz = n.ghG[o + 256] + n.bhhG[k + 256];
      float hn = n.ghG[o + 512] + n.bhhG[k + 512];
      float r_ = sigmoidf_(ir + hr);
      float z_ = sigmoidf_(iz + hz);
      float nn = tanhf(in_ + r_ * hn);
      float h = n.g_prev[i];
      float outv = (1.f - z_) * nn + z_ * h;
      n.g_prev[i] = outv;
      n.g_prevH[i] = (_Float16)outv;
      n.g_buf[((long long)t * kB + b) * kDG + k] = outv;
    }
    // ---- p GRU combine (q_new) ----
    for (int i = tid; i < kB * kP * kDP; i += blockDim.x) {
      int row = i >> 8, k = i & 255;
      int b = row >> 1, p = row & 1;
      long long gio = ((long long)t * kB + b) * k3H;
      long long co = (long long)b * k3H + k;
      long long ho = (long long)row * k3H + k;
      float ir  = (float)n.piU[gio + k]       + n.giC[co]       + n.bihP[k];
      float iz  = (float)n.piU[gio + 256 + k] + n.giC[co + 256] + n.bihP[k + 256];
      float in_ = (float)n.piU[gio + 512 + k] + n.giC[co + 512] + n.bihP[k + 512];
      float hr = n.ghP[ho]       + n.bhhP[k];
      float hz = n.ghP[ho + 256] + n.bhhP[k + 256];
      float hn = n.ghP[ho + 512] + n.bhhP[k + 512];
      float r_ = sigmoidf_(ir + hr);
      float z_ = sigmoidf_(iz + hz);
      float nn = tanhf(in_ + r_ * hn);
      float h = n.q[i];
      float qs = (1.f - z_) * nn + z_ * h;
      if (p == qit[b]) {
        n.q[i] = qs;
        n.qH[i] = (_Float16)qs;
        n.qselnewH[b * kDP + k] = (_Float16)qs;
      }
    }
    __syncthreads();

    // ---- e GEMMs ----
    block_gemm_rc(n.qselnewH, kDP, kB, n.WihE, kDP, n.giE, wid, lane);
    block_gemm_rc(n.e_prevH,  kDE, kB, n.WhhE, kDE, n.ghE, wid, lane);
    __syncthreads();

    // ---- e GRU combine ----
    for (int i = tid; i < kB * kDE; i += blockDim.x) {
      int b = i >> 8, k = i & 255;
      long long o = (long long)b * k3H + k;
      float ir  = n.giE[o]       + n.bihE[k];
      float iz  = n.giE[o + 256] + n.bihE[k + 256];
      float in_ = n.giE[o + 512] + n.bihE[k + 512];
      float hr = n.ghE[o]       + n.bhhE[k];
      float hz = n.ghE[o + 256] + n.bhhE[k + 256];
      float hn = n.ghE[o + 512] + n.bhhE[k + 512];
      float r_ = sigmoidf_(ir + hr);
      float z_ = sigmoidf_(iz + hz);
      float nn = tanhf(in_ + r_ * hn);
      float h = n.e_prev[i];
      float outv = (1.f - z_) * nn + z_ * h;
      n.e_prev[i] = outv;
      n.e_prevH[i] = (_Float16)outv;
      n.es[((long long)t * kB + b) * kDE + k] = outv;
    }
    __syncthreads();
  }
}

// ---------------------------------------------------------------------------
// Host-side GEMM launch helper
// ---------------------------------------------------------------------------
static void gemm(hipStream_t s,
                 const _Float16* A, long long lda, long long Ab,
                 const _Float16* B, long long ldb, long long Bb,
                 float* C, long long ldc, long long Cb,
                 _Float16* C16, long long ldc16, long long C16b,
                 const float* bias, int relu,
                 int M, int N, int K, int batch) {
  GemmArgs g;
  g.A = A; g.lda = lda; g.Abatch = Ab;
  g.B = B; g.ldb = ldb; g.Bbatch = Bb;
  g.C = C; g.ldc = ldc; g.Cbatch = Cb;
  g.C16 = C16; g.ldc16 = ldc16; g.C16batch = C16b;
  g.bias = bias; g.M = M; g.N = N; g.K = K; g.relu = relu;
  long long tiles = (long long)(M / 16) * (N / 32);
  int blocks = (int)((tiles + 7) / 8);
  dim3 grid((unsigned)blocks, (unsigned)batch);
  k_gemm<<<grid, 256, 0, s>>>(g);
}

// ---------------------------------------------------------------------------
// kernel_launch
// ---------------------------------------------------------------------------
extern "C" void kernel_launch(void* const* d_in, const int* in_sizes, int n_in,
                              void* d_out, int out_size, void* d_ws, size_t ws_size,
                              hipStream_t stream) {
  (void)in_sizes; (void)n_in; (void)out_size; (void)ws_size;

  const float* U[4] = {(const float*)d_in[0], (const float*)d_in[1],
                       (const float*)d_in[2], (const float*)d_in[3]};
  const float* qmask = (const float*)d_in[4];
  const float* umask = (const float*)d_in[5];
  const float* linW = (const float*)d_in[84];
  const float* linB = (const float*)d_in[85];
  const float* fcW  = (const float*)d_in[86];
  const float* fcB  = (const float*)d_in[87];
  const int dmod[4] = {1024, 512, 512, 1024};   // U_t, U_a, U_v, U_context
  const int dmN[6]  = {1024, 1024, 512, 512, 512, 512};

  // bump allocator over workspace
  size_t off = 0;
  auto alloc = [&](size_t bytes) -> void* {
    void* p = (char*)d_ws + off;
    off += (bytes + 255) & ~(size_t)255;
    return p;
  };

  int* cb  = (int*)alloc(kB * sizeof(int));
  int* qif = (int*)alloc((size_t)kT * kB * sizeof(int));
  int* qir = (int*)alloc((size_t)kT * kB * sizeof(int));

  _Float16 *Uh[4], *Urh[4];
  for (int m = 0; m < 4; ++m) {
    Uh[m]  = (_Float16*)alloc((size_t)kT * kB * dmod[m] * 2);
    Urh[m] = (_Float16*)alloc((size_t)kT * kB * dmod[m] * 2);
  }

  struct NetW {
    _Float16 *WihG, *WhhG, *WihP, *WhhP, *WihE, *WhhE, *giU, *piU;
    float *gprev, *q, *eprev, *gbuf, *es;
    _Float16 *gprevH, *qH, *eprevH, *qselH, *qselnewH, *csumH;
    float *ghG, *giQ, *ghP, *giC, *giE, *ghE;
  } nw[6];
  for (int nn = 0; nn < 6; ++nn) {
    nw[nn].WihG = (_Float16*)alloc((size_t)k3H * (kDMC + kDP) * 2);
    nw[nn].WhhG = (_Float16*)alloc((size_t)k3H * kDG * 2);
    nw[nn].WihP = (_Float16*)alloc((size_t)k3H * (dmN[nn] + kDG) * 2);
    nw[nn].WhhP = (_Float16*)alloc((size_t)k3H * kDP * 2);
    nw[nn].WihE = (_Float16*)alloc((size_t)k3H * kDP * 2);
    nw[nn].WhhE = (_Float16*)alloc((size_t)k3H * kDE * 2);
    nw[nn].giU  = (_Float16*)alloc((size_t)kT * kB * k3H * 2);
    nw[nn].piU  = (_Float16*)alloc((size_t)kT * kB * k3H * 2);
    nw[nn].gprev = (float*)alloc((size_t)kB * kDG * 4);
    nw[nn].q     = (float*)alloc((size_t)kB * kP * kDP * 4);
    nw[nn].eprev = (float*)alloc((size_t)kB * kDE * 4);
    nw[nn].gbuf  = (float*)alloc((size_t)kT * kB * kDG * 4);
    nw[nn].es    = (float*)alloc((size_t)kT * kB * kDE * 4);
    nw[nn].gprevH   = (_Float16*)alloc((size_t)kB * kDG * 2);
    nw[nn].qH       = (_Float16*)alloc((size_t)kB * kP * kDP * 2);
    nw[nn].eprevH   = (_Float16*)alloc((size_t)kB * kDE * 2);
    nw[nn].qselH    = (_Float16*)alloc((size_t)kB * kDP * 2);
    nw[nn].qselnewH = (_Float16*)alloc((size_t)kB * kDP * 2);
    nw[nn].csumH    = (_Float16*)alloc((size_t)kB * kDG * 2);
    nw[nn].ghG = (float*)alloc((size_t)kB * k3H * 4);
    nw[nn].giQ = (float*)alloc((size_t)kB * k3H * 4);
    nw[nn].ghP = (float*)alloc((size_t)kB * kP * k3H * 4);
    nw[nn].giC = (float*)alloc((size_t)kB * k3H * 4);
    nw[nn].giE = (float*)alloc((size_t)kB * k3H * 4);
    nw[nn].ghE = (float*)alloc((size_t)kB * k3H * 4);
  }
  _Float16* linWh = (_Float16*)alloc((size_t)512 * 512 * 2);
  _Float16* emh[3]; float* hM[3]; _Float16* hMh[3]; _Float16* hT[3];
  for (int m = 0; m < 3; ++m) {
    emh[m] = (_Float16*)alloc((size_t)kT * kB * 512 * 2);
    hM[m]  = (float*)alloc((size_t)kT * kB * 512 * 4);
    hMh[m] = (_Float16*)alloc((size_t)kT * kB * 512 * 2);
    hT[m]  = (_Float16*)alloc((size_t)kB * 512 * kT * 2);
  }
  float* scBB    = (float*)alloc((size_t)kB * kT * kT * 4);
  _Float16* nHh  = (_Float16*)alloc((size_t)kB * kT * kT * 2);
  float* a1f = (float*)alloc((size_t)kB * kT * 512 * 4);
  float* a2f = (float*)alloc((size_t)kB * kT * 512 * 4);
  float* hbi[3];
  for (int m = 0; m < 3; ++m) hbi[m] = (float*)alloc((size_t)kT * kB * 1024 * 4);

  // ---- preprocessing ----
  k_counts<<<1, kB, 0, stream>>>(umask, cb);
  k_qi<<<(kT * kB + 255) / 256, 256, 0, stream>>>(qmask, cb, qif, qir);
  for (int m = 0; m < 4; ++m) {
    long long nel = (long long)kT * kB * dmod[m];
    k_cvt<<<(int)((nel + 255) / 256), 256, 0, stream>>>(U[m], Uh[m], nel);
    k_reverse_f16<<<(int)((nel + 255) / 256), 256, 0, stream>>>(U[m], Urh[m], cb, dmod[m]);
  }
  k_cvt<<<(512 * 512 + 255) / 256, 256, 0, stream>>>(linW, linWh, 512 * 512);

  for (int nn = 0; nn < 6; ++nn) {
    auto cv = [&](int idx, _Float16* dst, long long nel) {
      k_cvt<<<(int)((nel + 255) / 256), 256, 0, stream>>>((const float*)d_in[idx], dst, nel);
    };
    int b0 = 6 + nn * 13;
    cv(b0 + 0, nw[nn].WihG, (long long)k3H * (kDMC + kDP));
    cv(b0 + 1, nw[nn].WhhG, (long long)k3H * kDG);
    cv(b0 + 4, nw[nn].WihP, (long long)k3H * (dmN[nn] + kDG));
    cv(b0 + 5, nw[nn].WhhP, (long long)k3H * kDP);
    cv(b0 + 8, nw[nn].WihE, (long long)k3H * kDP);
    cv(b0 + 9, nw[nn].WhhE, (long long)k3H * kDE);
  }

  // ---- big parallel input-projection GEMMs (bulk of FLOPs, WMMA) ----
  for (int nn = 0; nn < 6; ++nn) {
    int mi = nn >> 1;
    bool rev = nn & 1;
    const _Float16* Ain  = rev ? Urh[mi] : Uh[mi];
    const _Float16* AinC = rev ? Urh[3]  : Uh[3];
    gemm(stream, AinC, kDMC, 0, nw[nn].WihG, kDMC + kDP, 0,
         nullptr, 0, 0, nw[nn].giU, k3H, 0, nullptr, 0,
         kT * kB, k3H, kDMC, 1);
    gemm(stream, Ain, dmN[nn], 0, nw[nn].WihP, dmN[nn] + kDG, 0,
         nullptr, 0, 0, nw[nn].piU, k3H, 0, nullptr, 0,
         kT * kB, k3H, dmN[nn], 1);
  }

  // ---- persistent recurrent scan: 6 nets in parallel ----
  AllCtx ac;
  for (int nn = 0; nn < 6; ++nn) {
    int b0 = 6 + nn * 13;
    NetCtx& c = ac.net[nn];
    c.WihG = nw[nn].WihG; c.WhhG = nw[nn].WhhG;
    c.WihP = nw[nn].WihP; c.WhhP = nw[nn].WhhP;
    c.WihE = nw[nn].WihE; c.WhhE = nw[nn].WhhE;
    c.bihG = (const float*)d_in[b0 + 2];  c.bhhG = (const float*)d_in[b0 + 3];
    c.bihP = (const float*)d_in[b0 + 6];  c.bhhP = (const float*)d_in[b0 + 7];
    c.bihE = (const float*)d_in[b0 + 10]; c.bhhE = (const float*)d_in[b0 + 11];
    c.attnW = (const float*)d_in[b0 + 12];
    c.giU = nw[nn].giU; c.piU = nw[nn].piU;
    c.qi = (nn & 1) ? qir : qif;
    c.g_prev = nw[nn].gprev; c.q = nw[nn].q; c.e_prev = nw[nn].eprev;
    c.g_prevH = nw[nn].gprevH; c.qH = nw[nn].qH; c.e_prevH = nw[nn].eprevH;
    c.qselH = nw[nn].qselH; c.qselnewH = nw[nn].qselnewH; c.csumH = nw[nn].csumH;
    c.g_buf = nw[nn].gbuf; c.es = nw[nn].es;
    c.ghG = nw[nn].ghG; c.giQ = nw[nn].giQ; c.ghP = nw[nn].ghP;
    c.giC = nw[nn].giC; c.giE = nw[nn].giE; c.ghE = nw[nn].ghE;
    c.dmP = dmN[nn];
  }
  k_recurrent<<<6, 512, 0, stream>>>(ac);

  // ---- em = concat(ef, reverse(eb)); h = relu(em @ linW^T + b); transpose ----
  long long nel512 = (long long)kT * kB * 512;
  for (int m = 0; m < 3; ++m) {
    k_build_em<<<(int)((nel512 + 255) / 256), 256, 0, stream>>>(
        nw[2 * m].es, nw[2 * m + 1].es, cb, emh[m]);
    gemm(stream, emh[m], 512, 0, linWh, 512, 0,
         hM[m], 512, 0, hMh[m], 512, 0, linB, 1,
         kT * kB, 512, 512, 1);
    k_transpose_h<<<(int)((nel512 + 255) / 256), 256, 0, stream>>>(hM[m], hT[m]);
  }

  // ---- bimodal cross attention (batched WMMA, all operands A*W^T form) ----
  auto bimodal = [&](int i1, int i2, float* outp) {
    gemm(stream, hMh[i1], (long long)kB * 512, 512,
         hMh[i2], (long long)kB * 512, 512,
         scBB, kT, (long long)kT * kT, nullptr, 0, 0, nullptr, 0,
         kT, kT, 512, kB);
    k_softmax_f16<<<(kB * kT + 7) / 8, 256, 0, stream>>>(scBB, nHh, kB * kT, kT);
    gemm(stream, nHh, kT, (long long)kT * kT,
         hT[i2], kT, (long long)512 * kT,
         a1f, 512, (long long)kT * 512, nullptr, 0, 0, nullptr, 0,
         kT, 512, kT, kB);
    gemm(stream, hMh[i2], (long long)kB * 512, 512,
         hMh[i1], (long long)kB * 512, 512,
         scBB, kT, (long long)kT * kT, nullptr, 0, 0, nullptr, 0,
         kT, kT, 512, kB);
    k_softmax_f16<<<(kB * kT + 7) / 8, 256, 0, stream>>>(scBB, nHh, kB * kT, kT);
    gemm(stream, nHh, kT, (long long)kT * kT,
         hT[i1], kT, (long long)512 * kT,
         a2f, 512, (long long)kT * 512, nullptr, 0, 0, nullptr, 0,
         kT, 512, kT, kB);
    k_bicombine<<<(int)((nel512 + 255) / 256), 256, 0, stream>>>(
        a1f, a2f, hM[i1], hM[i2], outp);
  };
  bimodal(0, 1, hbi[0]);  // h_at
  bimodal(2, 1, hbi[1]);  // h_va
  bimodal(0, 2, hbi[2]);  // h_tv

  // ---- final FC + log_softmax ----
  k_fc<<<(kT * kB + 7) / 8, 256, 0, stream>>>(
      hbi[0], hbi[1], hbi[2], hM[0], hM[1], hM[2], fcW, fcB, (float*)d_out);
}
